// VelocityGateModel_56109452755475
// MI455X (gfx1250) — matrix operand order repaired and checked by reading, not compiled, not run on hardware
//
#include <hip/hip_runtime.h>
#include <hip/hip_bf16.h>

// VelocityGateModel forward for MI455X (gfx1250, wave32, WMMA).
// GEMMs: v_wmma_f32_16x16x32_f16, weights pre-transposed to [N x K] f16 so
// both operands load as contiguous 16B vectors. Generic GEMMs use a 16x64
// output tile per wave (4 independent accumulators -> A-operand register
// reuse and no WMMA->WMMA RAW hazard padding).
// Sequential velocity-gate scan keeps the 128x128 memory matrix in registers.

typedef __attribute__((ext_vector_type(16))) _Float16 v16h;
typedef __attribute__((ext_vector_type(8)))  _Float16 v8h;
typedef __attribute__((ext_vector_type(8)))  float    v8f;

#define BB   64
#define LL   512
#define HD   128
#define H2D  256
#define VV   32000
#define NTOK (BB * LL)

// ---------------------------------------------------------------------------
// Single 16x16 tile (used by the fused LN kernel).
// A row-major [M x K] f16; Wt row-major [N x K] f16. K multiple of 32.
// ---------------------------------------------------------------------------
static __device__ inline v8f wmma_tile_f16(const _Float16* __restrict__ A, int lda,
                                           const _Float16* __restrict__ Wt, int ldk,
                                           int row0, int col0, int K, int lane) {
  v8f acc;
#pragma unroll
  for (int i = 0; i < 8; ++i) acc[i] = 0.0f;
  const int m   = lane & 15;
  const int kbA = (lane < 16) ? 0 : 8;
  const int n   = lane & 15;
  const int kbB = (lane < 16) ? 0 : 8;
  const _Float16* ap = A  + (size_t)(row0 + m) * lda + kbA;
  const _Float16* bp = Wt + (size_t)(col0 + n) * ldk + kbB;
  for (int k0 = 0; k0 < K; k0 += 32) {
    v8h alo = *reinterpret_cast<const v8h*>(ap + k0);
    v8h ahi = *reinterpret_cast<const v8h*>(ap + k0 + 16);
    v8h blo = *reinterpret_cast<const v8h*>(bp + k0);
    v8h bhi = *reinterpret_cast<const v8h*>(bp + k0 + 16);
    v16h a, b;
#pragma unroll
    for (int e = 0; e < 8; ++e) {
      a[e] = alo[e]; a[8 + e] = ahi[e];
      b[e] = blo[e]; b[8 + e] = bhi[e];
    }
    acc = __builtin_amdgcn_wmma_f32_16x16x32_f16(false, a, false, b,
                                                 (short)0, acc, false, false);
  }
  return acc;
}

// ---------------------------------------------------------------------------
// 16x64 tile per wave: 4 independent accumulators, A registers reused 4x.
// ---------------------------------------------------------------------------
static __device__ inline void wmma_tile4_f16(const _Float16* __restrict__ A, int lda,
                                             const _Float16* __restrict__ Wt, int ldk,
                                             int row0, int col0, int K, int lane,
                                             v8f acc[4]) {
#pragma unroll
  for (int c = 0; c < 4; ++c)
#pragma unroll
    for (int i = 0; i < 8; ++i) acc[c][i] = 0.0f;
  const int m   = lane & 15;
  const int kbA = (lane < 16) ? 0 : 8;
  const int n   = lane & 15;
  const int kbB = (lane < 16) ? 0 : 8;
  const _Float16* ap = A  + (size_t)(row0 + m) * lda + kbA;
  const _Float16* bp = Wt + (size_t)(col0 + n) * ldk + kbB;
  for (int k0 = 0; k0 < K; k0 += 32) {
    v8h alo = *reinterpret_cast<const v8h*>(ap + k0);
    v8h ahi = *reinterpret_cast<const v8h*>(ap + k0 + 16);
    v16h a;
#pragma unroll
    for (int e = 0; e < 8; ++e) { a[e] = alo[e]; a[8 + e] = ahi[e]; }
    v16h b[4];
#pragma unroll
    for (int c = 0; c < 4; ++c) {
      const _Float16* bc = bp + (size_t)(c * 16) * ldk + k0;
      v8h blo = *reinterpret_cast<const v8h*>(bc);
      v8h bhi = *reinterpret_cast<const v8h*>(bc + 16);
#pragma unroll
      for (int e = 0; e < 8; ++e) { b[c][e] = blo[e]; b[c][8 + e] = bhi[e]; }
    }
#pragma unroll
    for (int c = 0; c < 4; ++c)
      acc[c] = __builtin_amdgcn_wmma_f32_16x16x32_f16(false, a, false, b[c],
                                                      (short)0, acc[c], false, false);
  }
}

// ---------------------------------------------------------------------------
// Transpose + convert: src f32 [K x N] row-major -> dst f16 [N x K] row-major
// ---------------------------------------------------------------------------
__global__ __launch_bounds__(256) void vg_cvt_t_f16(const float* __restrict__ src,
                                                    _Float16* __restrict__ dst,
                                                    int K, int N) {
  int i = blockIdx.x * 256 + threadIdx.x;   // coalesced read
  if (i < K * N) {
    int k = i / N, n = i % N;
    dst[(size_t)n * K + k] = (_Float16)src[i];
  }
}

// ---------------------------------------------------------------------------
// Embedding gather: h[t,:] = embed[seq[t],:] (f32 + f16 copies)
// ---------------------------------------------------------------------------
__global__ __launch_bounds__(256) void vg_gather(const int* __restrict__ seq,
                                                 const float* __restrict__ embed,
                                                 float* __restrict__ h,
                                                 _Float16* __restrict__ h16) {
  int i = blockIdx.x * 256 + threadIdx.x;   // over NTOK*HD
  int t = i >> 7, j = i & 127;
  float v = embed[(size_t)seq[t] * HD + j];
  h[i] = v;
  h16[i] = (_Float16)v;
}

// ---------------------------------------------------------------------------
// Generic WMMA GEMM, f16 output, bias + optional relu. Wt is [N x K].
// 16x64 output tile per wave; tile = blockIdx.x*8 + wave over (M/16)*(N/64).
// ---------------------------------------------------------------------------
__global__ __launch_bounds__(256) void vg_gemm_f16out(const _Float16* __restrict__ A,
                                                      const _Float16* __restrict__ Wt,
                                                      const float* __restrict__ bias,
                                                      _Float16* __restrict__ Out,
                                                      int M, int N, int K, int relu) {
  const int lane = threadIdx.x & 31;
  const int wave = threadIdx.x >> 5;
  const int ngrp = N >> 6;                  // groups of 64 columns
  const int tile = blockIdx.x * 8 + wave;
  if (tile >= (M >> 4) * ngrp) return;      // wave-uniform
  const int mt = tile / ngrp, ng = tile % ngrp;
  v8f acc[4];
  wmma_tile4_f16(A, K, Wt, K, mt * 16, ng * 64, K, lane, acc);
  const int n = lane & 15;
  const int mo = (lane < 16) ? 0 : 8;
#pragma unroll
  for (int c = 0; c < 4; ++c) {
    const int col = ng * 64 + c * 16 + n;
    const float bv = bias[col];
#pragma unroll
    for (int r = 0; r < 8; ++r) {
      float x = acc[c][r] + bv;
      if (relu) x = x > 0.0f ? x : 0.0f;
      Out[(size_t)(mt * 16 + mo + r) * N + col] = (_Float16)x;
    }
  }
}

// ---------------------------------------------------------------------------
// Generic WMMA GEMM, f32 output, optional bias (nullptr -> none). Wt is [N x K].
// ---------------------------------------------------------------------------
__global__ __launch_bounds__(256) void vg_gemm_f32out(const _Float16* __restrict__ A,
                                                      const _Float16* __restrict__ Wt,
                                                      const float* __restrict__ bias,
                                                      float* __restrict__ Out,
                                                      int M, int N, int K) {
  const int lane = threadIdx.x & 31;
  const int wave = threadIdx.x >> 5;
  const int ngrp = N >> 6;
  const int tile = blockIdx.x * 8 + wave;
  if (tile >= (M >> 4) * ngrp) return;
  const int mt = tile / ngrp, ng = tile % ngrp;
  v8f acc[4];
  wmma_tile4_f16(A, K, Wt, K, mt * 16, ng * 64, K, lane, acc);
  const int n = lane & 15;
  const int mo = (lane < 16) ? 0 : 8;
#pragma unroll
  for (int c = 0; c < 4; ++c) {
    const int col = ng * 64 + c * 16 + n;
    const float bv = bias ? bias[col] : 0.0f;
#pragma unroll
    for (int r = 0; r < 8; ++r)
      Out[(size_t)(mt * 16 + mo + r) * N + col] = acc[c][r] + bv;
  }
}

// ---------------------------------------------------------------------------
// GEMM2 (A1[NTOKx256] @ W2[256x128]) + b2 + residual + LayerNorm -> f16.
// Block owns a full 16x128 row slab: 8 waves, wave w = column tile w.
// ---------------------------------------------------------------------------
__global__ __launch_bounds__(256) void vg_gemm2_res_ln(const _Float16* __restrict__ A1,
                                                       const _Float16* __restrict__ W2t,
                                                       const float* __restrict__ b2,
                                                       const float* __restrict__ hres,
                                                       const float* __restrict__ gamma,
                                                       const float* __restrict__ beta,
                                                       _Float16* __restrict__ HN) {
  __shared__ float tile[16][HD + 1];
  const int lane = threadIdx.x & 31;
  const int wave = threadIdx.x >> 5;
  const int row0 = blockIdx.x * 16;
  v8f acc = wmma_tile_f16(A1, H2D, W2t, H2D, row0, wave * 16, H2D, lane);
  const int n = lane & 15;
  const int mo = (lane < 16) ? 0 : 8;
  const int col = wave * 16 + n;
  const float bv = b2[col];
#pragma unroll
  for (int r = 0; r < 8; ++r)
    tile[mo + r][col] = acc[r] + bv + hres[(size_t)(row0 + mo + r) * HD + col];
  __syncthreads();
  if (threadIdx.x < 16) {
    const int r = threadIdx.x;
    float s = 0.0f, s2 = 0.0f;
#pragma unroll 4
    for (int j = 0; j < HD; ++j) { float x = tile[r][j]; s += x; s2 += x * x; }
    const float mu  = s * (1.0f / HD);
    const float var = s2 * (1.0f / HD) - mu * mu;
    const float inv = rsqrtf(var + 1e-5f);
#pragma unroll 4
    for (int j = 0; j < HD; ++j)
      HN[(size_t)(row0 + r) * HD + j] =
          (_Float16)((tile[r][j] - mu) * inv * gamma[j] + beta[j]);
  }
}

// ---------------------------------------------------------------------------
// Sequential velocity-gate scan. One block per batch; M (128x128) lives in
// registers: thread tid holds row (tid&127), columns (tid>>7)*64 .. +63.
// ---------------------------------------------------------------------------
static __device__ inline float vg_blk_sum(float x, int tid, float* red) {
#pragma unroll
  for (int off = 16; off > 0; off >>= 1) x += __shfl_xor(x, off, 32);
  __syncthreads();                 // protect red from previous use
  if ((tid & 31) == 0) red[tid >> 5] = x;
  __syncthreads();
  float s = 0.0f;
#pragma unroll
  for (int w = 0; w < 8; ++w) s += red[w];
  return s;
}

__global__ __launch_bounds__(256) void vg_scan(const float* __restrict__ k_all,
                                               float* __restrict__ readv,
                                               _Float16* __restrict__ read16) {
  __shared__ float kv[HD], kn[HD], vp[HD], vp_prev[HD];
  __shared__ float partial[256];
  __shared__ float red[8];
  const int b = blockIdx.x;
  const int tid = threadIdx.x;
  const int row = tid & 127;
  const int colbase = (tid >> 7) * 64;
  float Mreg[64];
#pragma unroll
  for (int j = 0; j < 64; ++j) Mreg[j] = 0.0f;
  if (tid < HD) vp_prev[tid] = 0.0f;
  __syncthreads();
  const float* kb = k_all + (size_t)b * LL * HD;

  for (int t = 0; t < LL - 1; ++t) {
    if (tid < HD) kv[tid] = kb[(size_t)t * HD + tid];
    __syncthreads();
    float x = (tid < HD) ? kv[tid] * kv[tid] : 0.0f;
    const float kn2 = vg_blk_sum(x, tid, red);
    const float knorm = sqrtf(kn2);
    if (tid < HD) kn[tid] = kv[tid] / fmaxf(knorm, 1e-12f);
    __syncthreads();
    // vp = M @ kn  (each thread: 64-wide partial dot over its register slice)
    float dot = 0.0f;
#pragma unroll
    for (int j = 0; j < 64; ++j) dot += Mreg[j] * kn[colbase + j];
    partial[tid] = dot;
    __syncthreads();
    if (tid < HD) vp[tid] = partial[tid] + partial[tid + 128];
    __syncthreads();
    float dv = (tid < HD) ? (vp[tid] - vp_prev[tid]) : 0.0f;
    const float vel2 = vg_blk_sum(dv * dv, tid, red);
    const float vel = sqrtf(vel2) / fmaxf(knorm, 1e-6f);
    const float fire = (vel >= 0.1f) ? 0.05f : 0.0f;   // EMA_SCALE * fire
    if (fire != 0.0f) {
      const float coef = fire * (kv[row] - vp[row]);
#pragma unroll
      for (int j = 0; j < 64; ++j) Mreg[j] += coef * kn[colbase + j];
    }
    if (tid < HD) vp_prev[tid] = vp[tid];
    __syncthreads();
  }

  // read = M @ q, q = k_all[b, L-1, :]
  if (tid < HD) kv[tid] = kb[(size_t)(LL - 1) * HD + tid];
  __syncthreads();
  float dot = 0.0f;
#pragma unroll
  for (int j = 0; j < 64; ++j) dot += Mreg[j] * kv[colbase + j];
  partial[tid] = dot;
  __syncthreads();
  if (tid < HD) {
    const float rv = partial[tid] + partial[tid + 128];
    readv[(size_t)b * HD + tid] = rv;
    read16[(size_t)b * HD + tid] = (_Float16)rv;
  }
}

// ---------------------------------------------------------------------------
// Launch
// ---------------------------------------------------------------------------
extern "C" void kernel_launch(void* const* d_in, const int* in_sizes, int n_in,
                              void* d_out, int out_size, void* d_ws, size_t ws_size,
                              hipStream_t stream) {
  const int*   seq   = (const int*)  d_in[0];
  const float* embed = (const float*)d_in[1];
  const float* W1    = (const float*)d_in[2];
  const float* b1    = (const float*)d_in[3];
  const float* W2    = (const float*)d_in[4];
  const float* b2    = (const float*)d_in[5];
  const float* gamma = (const float*)d_in[6];
  const float* beta  = (const float*)d_in[7];
  const float* Wk    = (const float*)d_in[8];
  const float* Wr    = (const float*)d_in[9];
  const float* br    = (const float*)d_in[10];
  const float* Wo    = (const float*)d_in[11];
  const float* bo    = (const float*)d_in[12];
  float* out = (float*)d_out;

  char* w = (char*)d_ws;
  size_t off = 0;
  auto alloc = [&](size_t bytes) -> char* {
    char* p = w + off;
    off += (bytes + 255) & ~(size_t)255;
    return p;
  };
  float*    hbuf   = (float*)   alloc((size_t)NTOK * HD * 4);
  _Float16* h16    = (_Float16*)alloc((size_t)NTOK * HD * 2);
  _Float16* a1     = (_Float16*)alloc((size_t)NTOK * H2D * 2);
  _Float16* hn16   = (_Float16*)alloc((size_t)NTOK * HD * 2);
  float*    kall   = (float*)   alloc((size_t)NTOK * HD * 4);
  _Float16* W1t    = (_Float16*)alloc((size_t)HD * H2D * 2);   // [256 x 128]
  _Float16* W2t    = (_Float16*)alloc((size_t)H2D * HD * 2);   // [128 x 256]
  _Float16* Wkt    = (_Float16*)alloc((size_t)HD * HD * 2);    // [128 x 128]
  _Float16* Wrt    = (_Float16*)alloc((size_t)HD * HD * 2);    // [128 x 128]
  _Float16* Wot    = (_Float16*)alloc((size_t)HD * VV * 2);    // [32000 x 128]
  float*    readv  = (float*)   alloc((size_t)BB * HD * 4);
  _Float16* read16 = (_Float16*)alloc((size_t)BB * HD * 2);
  _Float16* r2h    = (_Float16*)alloc((size_t)BB * HD * 2);

  // Weight transpose + convert f32 -> f16 ([K x N] -> [N x K])
  vg_cvt_t_f16<<<(HD * H2D + 255) / 256, 256, 0, stream>>>(W1, W1t, HD, H2D);
  vg_cvt_t_f16<<<(H2D * HD + 255) / 256, 256, 0, stream>>>(W2, W2t, H2D, HD);
  vg_cvt_t_f16<<<(HD * HD + 255) / 256, 256, 0, stream>>>(Wk, Wkt, HD, HD);
  vg_cvt_t_f16<<<(HD * HD + 255) / 256, 256, 0, stream>>>(Wr, Wrt, HD, HD);
  vg_cvt_t_f16<<<(HD * VV + 255) / 256, 256, 0, stream>>>(Wo, Wot, HD, VV);

  // 1) embedding gather
  vg_gather<<<(NTOK * HD) / 256, 256, 0, stream>>>(seq, embed, hbuf, h16);

  // 2) MLP: a1 = relu(h @ W1 + b1)   [NTOK x 256]
  {
    int tiles = (NTOK / 16) * (H2D / 64);
    vg_gemm_f16out<<<(tiles + 7) / 8, 256, 0, stream>>>(h16, W1t, b1, a1,
                                                        NTOK, H2D, HD, 1);
  }
  // 3) h = LN(h + a1 @ W2 + b2) -> hn16
  vg_gemm2_res_ln<<<NTOK / 16, 256, 0, stream>>>(a1, W2t, b2, hbuf, gamma, beta, hn16);

  // 4) k_all = hn @ Wk   [NTOK x 128] f32
  {
    int tiles = (NTOK / 16) * (HD / 64);
    vg_gemm_f32out<<<(tiles + 7) / 8, 256, 0, stream>>>(hn16, Wkt, nullptr, kall,
                                                        NTOK, HD, HD);
  }
  // 5) sequential velocity-gate scan + read = M @ q
  vg_scan<<<BB, 256, 0, stream>>>(kall, readv, read16);

  // 6) r2 = read @ Wr + br   [64 x 128] f16
  {
    int tiles = (BB / 16) * (HD / 64);
    vg_gemm_f16out<<<(tiles + 7) / 8, 256, 0, stream>>>(read16, Wrt, br, r2h,
                                                        BB, HD, HD, 0);
  }
  // 7) out = r2 @ Wo + bo   [64 x 32000] f32
  {
    int tiles = (BB / 16) * (VV / 64);
    vg_gemm_f32out<<<(tiles + 7) / 8, 256, 0, stream>>>(r2h, Wot, bo, out,
                                                        BB, VV, HD);
  }
}
